// ViG_17033840296557
// MI455X (gfx1250) — compile-verified, matrix operand order checked by reading, not run on hardware
//
#include <hip/hip_runtime.h>
#include <hip/hip_bf16.h>

// ---------------------------------------------------------------------------
// ViG forward on gfx1250 (MI455X).
//  - Token GEMMs: V_WMMA_F32_16X16X32_F16, one wave per 16x16 tile,
//    4 waves/block share a 16xK f16 weight panel staged in LDS (ds_load_b128),
//    A streamed as pre-converted f16 (global_load_b128) + global_prefetch_b8.
//  - Convs / BN stats / KNN / top-k / aggregation: f32 VALU.
// ---------------------------------------------------------------------------

typedef __attribute__((ext_vector_type(16))) _Float16 v16h;
typedef __attribute__((ext_vector_type(8)))  _Float16 v8h;
typedef __attribute__((ext_vector_type(8)))  float    v8f;

#define ROWS   6272      // 32 images * 196 tokens
#define NTOK   196
#define BATCH  32
#define CC     192
#define KMAX   18
#define BN_EPS 1e-5f

// ------------------------------ WMMA GEMM ----------------------------------
// C[M,N] = A[M,K] (f16 row-major) * Bh[N,K]^T (f16 row-major, torch weight
// layout) + bias.  N%16==0, K%32==0.  Block = 128 thr (4 waves) = 64x16 tile.
__global__ __launch_bounds__(128)
void gemm_wmma_f16(const _Float16* __restrict__ A, const _Float16* __restrict__ Bh,
                   const float* __restrict__ bias, float* __restrict__ C,
                   int M, int N, int K)
{
    extern __shared__ _Float16 sB[];          // 16 x K weight panel
    const int tilesN = N >> 4;
    const int bM     = blockIdx.x / tilesN;   // 64-row macro tile index
    const int tn     = (blockIdx.x % tilesN) << 4;
    const int tid    = threadIdx.x;

    // cooperative stage of weight panel rows tn..tn+15 into LDS (b128 copies)
    const int chunksPerRow = K >> 3;          // v8h chunks per row
    for (int c = tid; c < 16 * chunksPerRow; c += 128) {
        int r  = c / chunksPerRow;
        int ck = (c % chunksPerRow) << 3;
        *(v8h*)(sB + r * K + ck) = *(const v8h*)(Bh + (size_t)(tn + r) * K + ck);
    }
    __syncthreads();

    const int wid = tid >> 5;
    const int tm  = (bM << 6) + (wid << 4);
    if (tm < M) {                              // wave-uniform; EXEC stays all-ones
        const int lane = tid & 31;
        const int half = lane >> 4;            // K sub-slice select (ISA A/B layout)
        const int l    = lane & 15;            // row (A) / col (B) within tile
        v8f acc = {};
        const _Float16* arow = A  + (size_t)(tm + l) * K + half * 8;
        const _Float16* brow = sB + l * K + half * 8;
        for (int k0 = 0; k0 < K; k0 += 32) {
            __builtin_prefetch(arow + k0 + 256, 0, 0);   // global_prefetch_b8
            v8h a0 = *(const v8h*)(arow + k0);
            v8h a1 = *(const v8h*)(arow + k0 + 16);
            v8h b0 = *(const v8h*)(brow + k0);           // ds_load_b128
            v8h b1 = *(const v8h*)(brow + k0 + 16);
            v16h a = __builtin_shufflevector(a0, a1, 0,1,2,3,4,5,6,7,8,9,10,11,12,13,14,15);
            v16h b = __builtin_shufflevector(b0, b1, 0,1,2,3,4,5,6,7,8,9,10,11,12,13,14,15);
            acc = __builtin_amdgcn_wmma_f32_16x16x32_f16(false, a, false, b,
                                                         (short)0, acc, false, false);
        }
        const int   col = tn + l;
        const float bv  = bias ? bias[col] : 0.0f;
        const int   r0  = tm + half * 8;
#pragma unroll
        for (int r = 0; r < 8; ++r)
            C[(size_t)(r0 + r) * N + col] = acc[r] + bv;
    }
}

// --------------------------- f32 -> f16 convert ----------------------------
__global__ void cvt_f16(const float* __restrict__ a, _Float16* __restrict__ o, int n)
{
    int i = blockIdx.x * blockDim.x + threadIdx.x;
    if (i < n) o[i] = (_Float16)a[i];
}

// ------------------------------ direct conv --------------------------------
__global__ void conv3x3_kernel(const float* __restrict__ x, const float* __restrict__ w,
                               const float* __restrict__ bias, float* __restrict__ y,
                               int B, int Cin, int Hin, int Win,
                               int Cout, int Hout, int Wout, int stride)
{
    size_t idx   = (size_t)blockIdx.x * blockDim.x + threadIdx.x;
    size_t total = (size_t)B * Cout * Hout * Wout;
    if (idx >= total) return;
    int wo = (int)(idx % Wout); size_t r = idx / Wout;
    int ho = (int)(r % Hout); r /= Hout;
    int co = (int)(r % Cout); int b = (int)(r / Cout);
    float s = bias[co];
    const float* wp = w + (size_t)co * Cin * 9;
    for (int ci = 0; ci < Cin; ++ci) {
        const float* xp = x + ((size_t)b * Cin + ci) * Hin * Win;
        const float* wc = wp + ci * 9;
        for (int kh = 0; kh < 3; ++kh) {
            int hi = ho * stride - 1 + kh;
            if (hi < 0 || hi >= Hin) continue;
            for (int kw = 0; kw < 3; ++kw) {
                int wi = wo * stride - 1 + kw;
                if (wi < 0 || wi >= Win) continue;
                s += xp[(size_t)hi * Win + wi] * wc[kh * 3 + kw];
            }
        }
    }
    y[idx] = s;
}

// ------------------------ BatchNorm2d (NCHW, batch stats) ------------------
__global__ __launch_bounds__(256)
void bn2d_stats(const float* __restrict__ x, int B, int C, int HW,
                float* __restrict__ mean, float* __restrict__ rinv)
{
    int c = blockIdx.x;
    float s = 0.f, s2 = 0.f;
    int total = B * HW;
    for (int i = threadIdx.x; i < total; i += 256) {
        int b = i / HW, p = i % HW;
        float v = x[((size_t)b * C + c) * HW + p];
        s += v; s2 += v * v;
    }
    __shared__ float sh[256], sh2[256];
    sh[threadIdx.x] = s; sh2[threadIdx.x] = s2;
    __syncthreads();
    for (int o = 128; o > 0; o >>= 1) {
        if (threadIdx.x < o) { sh[threadIdx.x] += sh[threadIdx.x + o];
                               sh2[threadIdx.x] += sh2[threadIdx.x + o]; }
        __syncthreads();
    }
    if (threadIdx.x == 0) {
        float inv = 1.0f / (float)total;
        float mu  = sh[0] * inv;
        float var = sh2[0] * inv - mu * mu;
        mean[c] = mu; rinv[c] = rsqrtf(var + BN_EPS);
    }
}

__global__ void bn2d_apply(float* __restrict__ x, size_t total, int C, int HW,
                           const float* __restrict__ mean, const float* __restrict__ rinv,
                           const float* __restrict__ g, const float* __restrict__ be, int relu)
{
    size_t i = (size_t)blockIdx.x * blockDim.x + threadIdx.x;
    if (i >= total) return;
    int c = (int)((i / HW) % C);
    float v = g[c] * (x[i] - mean[c]) * rinv[c] + be[c];
    if (relu) v = fmaxf(v, 0.f);
    x[i] = v;
}

// ---------------------- token BN (over batch+token dims) ------------------
__global__ __launch_bounds__(256)
void bn_col_stats(const float* __restrict__ x, int R, int N,
                  float* __restrict__ mean, float* __restrict__ rinv)
{
    int c = blockIdx.x;
    float s = 0.f, s2 = 0.f;
    for (int r = threadIdx.x; r < R; r += 256) {
        float v = x[(size_t)r * N + c];
        s += v; s2 += v * v;
    }
    __shared__ float sh[256], sh2[256];
    sh[threadIdx.x] = s; sh2[threadIdx.x] = s2;
    __syncthreads();
    for (int o = 128; o > 0; o >>= 1) {
        if (threadIdx.x < o) { sh[threadIdx.x] += sh[threadIdx.x + o];
                               sh2[threadIdx.x] += sh2[threadIdx.x + o]; }
        __syncthreads();
    }
    if (threadIdx.x == 0) {
        float inv = 1.0f / (float)R;
        float mu  = sh[0] * inv;
        float var = sh2[0] * inv - mu * mu;
        mean[c] = mu; rinv[c] = rsqrtf(var + BN_EPS);
    }
}

// normalize (+ReLU) (+residual); writes f32 out and optional f16 copy for the
// next WMMA GEMM.
__global__ void bn_col_apply(const float* __restrict__ x, float* __restrict__ out,
                             size_t total, int N,
                             const float* __restrict__ mean, const float* __restrict__ rinv,
                             const float* __restrict__ g, const float* __restrict__ be,
                             const float* __restrict__ residual, int relu,
                             _Float16* __restrict__ out16)
{
    size_t i = (size_t)blockIdx.x * blockDim.x + threadIdx.x;
    if (i >= total) return;
    int c = (int)(i % N);
    float v = g[c] * (x[i] - mean[c]) * rinv[c] + be[c];
    if (relu) v = fmaxf(v, 0.f);
    if (residual) v += residual[i];
    out[i] = v;
    if (out16) out16[i] = (_Float16)v;
}

// --------------------- pos-embed add + NCHW->token layout ------------------
__global__ void tokenize_addpos(const float* __restrict__ x, const float* __restrict__ pos,
                                float* __restrict__ t, _Float16* __restrict__ t16)
{
    size_t i = (size_t)blockIdx.x * blockDim.x + threadIdx.x;   // (b*196+n)*192+c
    if (i >= (size_t)ROWS * CC) return;
    int c = (int)(i % CC);
    int n = (int)((i / CC) % NTOK);
    int b = (int)(i / ((size_t)CC * NTOK));
    float v = x[((size_t)b * CC + c) * NTOK + n] + pos[c * NTOK + n];
    t[i]   = v;
    t16[i] = (_Float16)v;
}

// --------------------------- KNN graph build -------------------------------
__global__ __launch_bounds__(192)
void rownorm(const float* __restrict__ y, float* __restrict__ xn, float* __restrict__ sq)
{
    int row = blockIdx.x, c = threadIdx.x;
    float v = y[(size_t)row * CC + c];
    __shared__ float sh[256];
    sh[c] = v * v;
    if (c < 64) sh[192 + c] = 0.f;
    __syncthreads();
    for (int o = 128; o > 0; o >>= 1) {
        if (c < o) sh[c] += sh[c + o];
        __syncthreads();
    }
    float s  = sh[0];
    float rn = rsqrtf(s + 1e-12f);
    xn[(size_t)row * CC + c] = v * rn;
    if (c == 0) sq[row] = s * rn * rn;      // == sum(xn^2)
}

__global__ __launch_bounds__(196)
void knn_topk(const float* __restrict__ xn, const float* __restrict__ sq,
              int* __restrict__ idx, int k)
{
    int blk = blockIdx.x;                   // b*14 + rowgroup
    int b   = blk / 14;
    int r0  = (blk % 14) * 14;
    int tid = threadIdx.x;
    int nl  = tid / 14, ml = tid % 14;
    __shared__ float sA[14][CC];
    __shared__ float sD[14][NTOK];
    __shared__ float sQ[14];
    for (int i = tid; i < 14 * CC; i += 196) {
        int rr = i / CC, cc2 = i % CC;
        sA[rr][cc2] = xn[(size_t)(b * NTOK + r0 + rr) * CC + cc2];
    }
    if (tid < 14) sQ[tid] = sq[b * NTOK + r0 + tid];
    __syncthreads();
    for (int m0 = 0; m0 < NTOK; m0 += 14) {
        int m = m0 + ml;
        const float* xm = xn + (size_t)(b * NTOK + m) * CC;
        float dot = 0.f;
        for (int c2 = 0; c2 < CC; ++c2) dot += sA[nl][c2] * xm[c2];
        sD[nl][m] = sQ[nl] + sq[b * NTOK + m] - 2.0f * dot;
    }
    __syncthreads();
    if (tid < 14) {
        float bd[KMAX]; int bi[KMAX];
        for (int j = 0; j < k; ++j) { bd[j] = 3.0e38f; bi[j] = 0; }
        for (int m = 0; m < NTOK; ++m) {
            float d = sD[tid][m];
            if (d < bd[k - 1]) {
                int j = k - 1;
                while (j > 0 && bd[j - 1] > d) { bd[j] = bd[j - 1]; bi[j] = bi[j - 1]; --j; }
                bd[j] = d; bi[j] = m;
            }
        }
        int grow = b * NTOK + r0 + tid;
        for (int j = 0; j < k; ++j) idx[grow * KMAX + j] = bi[j];
    }
}

// max-relative aggregation + channel-interleave into [ROWS, 2C] (f32 + f16)
__global__ void aggregate(const float* __restrict__ y, const int* __restrict__ idx,
                          float* __restrict__ z, _Float16* __restrict__ z16, int k)
{
    size_t i = (size_t)blockIdx.x * blockDim.x + threadIdx.x;
    if (i >= (size_t)ROWS * CC) return;
    int c    = (int)(i % CC);
    int grow = (int)(i / CC);
    int b    = grow / NTOK;
    float self = y[i];
    float mx = -3.0e38f;
    int base = grow * KMAX;
    for (int j = 0; j < k; ++j) {
        int m = idx[base + j];
        float v = y[(size_t)(b * NTOK + m) * CC + c];
        mx = fmaxf(mx, v - self);
    }
    size_t o = (size_t)grow * (2 * CC) + 2 * c;
    z[o]     = self;  z16[o]     = (_Float16)self;
    z[o + 1] = mx;    z16[o + 1] = (_Float16)mx;
}

// -------------------------------- head -------------------------------------
__global__ void mean_pool(const float* __restrict__ t, float* __restrict__ pooled,
                          _Float16* __restrict__ pooled16)
{
    int i = blockIdx.x * blockDim.x + threadIdx.x;     // BATCH*CC
    if (i >= BATCH * CC) return;
    int c = i % CC, b = i / CC;
    float s = 0.f;
    for (int n = 0; n < NTOK; ++n) s += t[(size_t)(b * NTOK + n) * CC + c];
    float v = s * (1.0f / NTOK);
    pooled[i] = v;
    pooled16[i] = (_Float16)v;
}

__global__ void final_linear(const float* __restrict__ h, const float* __restrict__ w2,
                             const float* __restrict__ b2, float* __restrict__ out)
{
    int i = blockIdx.x * blockDim.x + threadIdx.x;     // 32*1000
    if (i >= BATCH * 1000) return;
    int o = i % 1000, b = i / 1000;
    const float* hp = h + (size_t)b * 1024;
    const float* wp = w2 + (size_t)o * 1024;
    float s = b2[o];
    for (int c = 0; c < 1024; ++c) s += hp[c] * wp[c];
    out[i] = s;
}

// ===========================================================================
extern "C" void kernel_launch(void* const* d_in, const int* in_sizes, int n_in,
                              void* d_out, int out_size, void* d_ws, size_t ws_size,
                              hipStream_t stream)
{
    if (n_in < 48 || !d_out || !d_ws) return;
    (void)in_sizes; (void)out_size; (void)ws_size;

    // ----- input map (pytree order of setup_inputs) -----
    const float* inp = (const float*)d_in[0];
    const float* pos = (const float*)d_in[1];
    const float *sw[5], *sb[5], *sg[5], *sbe[5];
    for (int i = 0; i < 5; ++i) {
        sw[i]  = (const float*)d_in[2 + 4 * i];
        sb[i]  = (const float*)d_in[3 + 4 * i];
        sg[i]  = (const float*)d_in[4 + 4 * i];
        sbe[i] = (const float*)d_in[5 + 4 * i];
    }
    const float* fc1_w  = (const float*)d_in[22];
    const float* fc1_b  = (const float*)d_in[23];
    const float* fc1_g  = (const float*)d_in[24];
    const float* fc1_be = (const float*)d_in[25];
    const float* mr_w   = (const float*)d_in[26];
    const float* mr_b   = (const float*)d_in[27];
    const float* mr_g   = (const float*)d_in[28];
    const float* mr_be  = (const float*)d_in[29];
    const float* fc2_w  = (const float*)d_in[30];
    const float* fc2_b  = (const float*)d_in[31];
    const float* fc2_g  = (const float*)d_in[32];
    const float* fc2_be = (const float*)d_in[33];
    const float* f1_w   = (const float*)d_in[34];
    const float* f1_b   = (const float*)d_in[35];
    const float* f1_g   = (const float*)d_in[36];
    const float* f1_be  = (const float*)d_in[37];
    const float* f2_w   = (const float*)d_in[38];
    const float* f2_b   = (const float*)d_in[39];
    const float* f2_g   = (const float*)d_in[40];
    const float* f2_be  = (const float*)d_in[41];
    const float* hw1 = (const float*)d_in[42];
    const float* hb1 = (const float*)d_in[43];
    const float* hg1 = (const float*)d_in[44];
    const float* hbe1= (const float*)d_in[45];
    const float* hw2 = (const float*)d_in[46];
    const float* hb2 = (const float*)d_in[47];

    // ----- workspace bump allocator -----
    size_t off = 0;
    auto wsalloc = [&](size_t bytes) -> void* {
        void* p = (char*)d_ws + off;
        off += (bytes + 255) & ~(size_t)255;
        return p;
    };
    float* bufP = (float*)wsalloc((size_t)9633792 * 4);   // conv ping (32*24*112*112)
    float* bufQ = (float*)wsalloc((size_t)4816896 * 4);   // conv pong
    float* T    = (float*)wsalloc((size_t)ROWS * CC * 4);
    float* Y    = (float*)wsalloc((size_t)ROWS * CC * 4);
    float* XN   = (float*)wsalloc((size_t)ROWS * CC * 4);
    float* SQ   = (float*)wsalloc((size_t)ROWS * 4);
    int*   IDX  = (int*)  wsalloc((size_t)ROWS * KMAX * 4);
    float* Z    = (float*)wsalloc((size_t)ROWS * 2 * CC * 4);
    float* G    = (float*)wsalloc((size_t)ROWS * CC * 4);
    float* H    = (float*)wsalloc((size_t)ROWS * 4 * CC * 4);
    float* MEAN = (float*)wsalloc(1024 * 4);
    float* RINV = (float*)wsalloc(1024 * 4);
    float* POOL = (float*)wsalloc((size_t)BATCH * CC * 4);
    float* HB   = (float*)wsalloc((size_t)BATCH * 1024 * 4);
    // f16 activation copies (WMMA A operands)
    _Float16* TH  = (_Float16*)wsalloc((size_t)ROWS * CC * 2);
    _Float16* ZH  = (_Float16*)wsalloc((size_t)ROWS * 2 * CC * 2);
    _Float16* GH  = (_Float16*)wsalloc((size_t)ROWS * CC * 2);
    _Float16* HH  = (_Float16*)wsalloc((size_t)ROWS * 4 * CC * 2);
    _Float16* PH  = (_Float16*)wsalloc((size_t)BATCH * CC * 2);
    // f16 weights (WMMA B operands)
    _Float16* fc1h = (_Float16*)wsalloc((size_t)12 * 192 * 192 * 2);
    _Float16* mrh  = (_Float16*)wsalloc((size_t)12 * 192 * 384 * 2);
    _Float16* fc2h = (_Float16*)wsalloc((size_t)12 * 192 * 192 * 2);
    _Float16* ff1h = (_Float16*)wsalloc((size_t)12 * 768 * 192 * 2);
    _Float16* ff2h = (_Float16*)wsalloc((size_t)12 * 192 * 768 * 2);
    _Float16* hdh  = (_Float16*)wsalloc((size_t)1024 * 192 * 2);

    // ----- weight f32->f16 conversion (recomputed every call; deterministic)
    auto cvt = [&](const float* s, _Float16* dh, int n) {
        cvt_f16<<<(unsigned)((n + 255) / 256), 256, 0, stream>>>(s, dh, n);
    };
    cvt(fc1_w, fc1h, 12 * 192 * 192);
    cvt(mr_w,  mrh,  12 * 192 * 384);
    cvt(fc2_w, fc2h, 12 * 192 * 192);
    cvt(f1_w,  ff1h, 12 * 768 * 192);
    cvt(f2_w,  ff2h, 12 * 192 * 768);
    cvt(hw1,   hdh,  1024 * 192);

    auto gemm = [&](const _Float16* A, const _Float16* Bh, const float* bias, float* C,
                    int M, int N, int K) {
        int blocks = ((M + 63) >> 6) * (N >> 4);
        size_t shmem = (size_t)16 * K * sizeof(_Float16);   // <= 24 KB
        gemm_wmma_f16<<<(unsigned)blocks, 128, shmem, stream>>>(A, Bh, bias, C, M, N, K);
    };
    auto stats = [&](const float* x, int R, int N) {
        bn_col_stats<<<(unsigned)N, 256, 0, stream>>>(x, R, N, MEAN, RINV);
    };
    auto apply = [&](const float* x, float* out, int R, int N,
                     const float* g, const float* be, const float* res, int relu,
                     _Float16* out16) {
        size_t tot = (size_t)R * N;
        bn_col_apply<<<(unsigned)((tot + 255) / 256), 256, 0, stream>>>(
            x, out, tot, N, MEAN, RINV, g, be, res, relu, out16);
    };

    // ----- stem: 5x conv3x3 + BN2d (+ReLU on first 4) -----
    struct LDesc { int ci, co, hi, wi, ho, wo, s; };
    const LDesc LS[5] = { {3,24,224,224,112,112,2}, {24,48,112,112,56,56,2},
                          {48,96,56,56,28,28,2},    {96,192,28,28,14,14,2},
                          {192,192,14,14,14,14,1} };
    const float* csrc = inp;
    float* cdst[5] = { bufP, bufQ, bufP, bufQ, bufP };
    for (int i = 0; i < 5; ++i) {
        const LDesc& L = LS[i];
        size_t tot = (size_t)BATCH * L.co * L.ho * L.wo;
        conv3x3_kernel<<<(unsigned)((tot + 255) / 256), 256, 0, stream>>>(
            csrc, sw[i], sb[i], cdst[i], BATCH, L.ci, L.hi, L.wi, L.co, L.ho, L.wo, L.s);
        bn2d_stats<<<(unsigned)L.co, 256, 0, stream>>>(cdst[i], BATCH, L.co, L.ho * L.wo,
                                                       MEAN, RINV);
        bn2d_apply<<<(unsigned)((tot + 255) / 256), 256, 0, stream>>>(
            cdst[i], tot, L.co, L.ho * L.wo, MEAN, RINV, sg[i], sbe[i], (i < 4) ? 1 : 0);
        csrc = cdst[i];
    }
    {   // x + pos_embed, NCHW -> [ROWS, C] tokens (f32 + f16)
        size_t tot = (size_t)ROWS * CC;
        tokenize_addpos<<<(unsigned)((tot + 255) / 256), 256, 0, stream>>>(bufP, pos, T, TH);
    }

    // ----- 12 Grapher + FFN blocks -----
    const int NUM_KNN[12] = { 9, 9, 10, 11, 12, 13, 13, 14, 15, 16, 17, 18 };
    const size_t tokTot = (size_t)ROWS * CC;
    for (int i = 0; i < 12; ++i) {
        int k = NUM_KNN[i];
        // Grapher: y = BN(fc1(t))
        gemm(TH, fc1h + (size_t)i * 192 * 192, fc1_b + i * 192, Y, ROWS, 192, 192);
        stats(Y, ROWS, 192);
        apply(Y, Y, ROWS, 192, fc1_g + i * 192, fc1_be + i * 192, nullptr, 0, nullptr);
        // knn graph (on L2-normalized y) + max-relative aggregation
        rownorm<<<ROWS, 192, 0, stream>>>(Y, XN, SQ);
        knn_topk<<<BATCH * 14, 196, 0, stream>>>(XN, SQ, IDX, k);
        aggregate<<<(unsigned)((tokTot + 255) / 256), 256, 0, stream>>>(Y, IDX, Z, ZH, k);
        // z -> ReLU(BN(mr(z)))
        gemm(ZH, mrh + (size_t)i * 192 * 384, mr_b + i * 192, G, ROWS, 192, 384);
        stats(G, ROWS, 192);
        apply(G, G, ROWS, 192, mr_g + i * 192, mr_be + i * 192, nullptr, 1, GH);
        // t = BN(fc2(.)) + t
        gemm(GH, fc2h + (size_t)i * 192 * 192, fc2_b + i * 192, Y, ROWS, 192, 192);
        stats(Y, ROWS, 192);
        apply(Y, T, ROWS, 192, fc2_g + i * 192, fc2_be + i * 192, T, 0, TH);
        // FFN: h = ReLU(BN(ffn1(t)))
        gemm(TH, ff1h + (size_t)i * 768 * 192, f1_b + i * 768, H, ROWS, 768, 192);
        stats(H, ROWS, 768);
        apply(H, H, ROWS, 768, f1_g + i * 768, f1_be + i * 768, nullptr, 1, HH);
        // t = BN(ffn2(h)) + t
        gemm(HH, ff2h + (size_t)i * 192 * 768, f2_b + i * 192, Y, ROWS, 192, 768);
        stats(Y, ROWS, 192);
        apply(Y, T, ROWS, 192, f2_g + i * 192, f2_be + i * 192, T, 0, TH);
    }

    // ----- head -----
    mean_pool<<<(unsigned)((BATCH * CC + 255) / 256), 256, 0, stream>>>(T, POOL, PH);
    gemm(PH, hdh, hb1, HB, BATCH, 1024, 192);
    stats(HB, BATCH, 1024);
    apply(HB, HB, BATCH, 1024, hg1, hbe1, nullptr, 1, nullptr);
    final_linear<<<(unsigned)((BATCH * 1000 + 255) / 256), 256, 0, stream>>>(
        HB, hw2, hb2, (float*)d_out);
}
